// RetrievalLoss_66314295050631
// MI455X (gfx1250) — compile-verified, matrix operand order checked by reading, not run on hardware
//
#include <hip/hip_runtime.h>

// ---------------------------------------------------------------------------
// RetrievalLoss (triplet mining) for MI455X / gfx1250.
// wave32 + WMMA bf16x3 split + async global->LDS double-buffered B staging.
// ---------------------------------------------------------------------------

#define NROWS 8192
#define DIM   256
#define NUM_KBLK (DIM / 32)   // 8 K-blocks of 32 for 16x16x32 bf16 WMMA
#define CSPLIT 4              // column chunks mined in parallel per row
#define CHUNK  (NROWS / CSPLIT)
#define MINE_WAVES 4          // rows per workgroup = MINE_WAVES * 16
#define MINE_THREADS (MINE_WAVES * 32)
#define TILE_BYTES (2 * 16 * DIM * 2)       // 2 planes * 16 rows * DIM bf16 = 16 KB
#define TILE_CHUNKS (TILE_BYTES / 16)       // 1024 b128 chunks
#define CHUNKS_PER_THREAD (TILE_CHUNKS / MINE_THREADS)  // 8
#define DELTA 1.0f

typedef __attribute__((ext_vector_type(8)))  __bf16 v8bf;
typedef __attribute__((ext_vector_type(16))) __bf16 v16bf;
typedef __attribute__((ext_vector_type(8)))  float  v8f;

__device__ __forceinline__ unsigned short f2bf_rn(float f) {
    unsigned int u = __float_as_uint(f);
    unsigned int r = u + 0x7FFFu + ((u >> 16) & 1u);   // round-to-nearest-even
    return (unsigned short)(r >> 16);
}
__device__ __forceinline__ float bf2f(unsigned short h) {
    return __uint_as_float(((unsigned int)h) << 16);
}

// ---------------------------------------------------------------------------
// Kernel 1: split fp32 into bf16 hi + bf16 lo residual.
// ---------------------------------------------------------------------------
__global__ void rl_split_kernel(const float* __restrict__ q,
                                unsigned short* __restrict__ qhi,
                                unsigned short* __restrict__ qlo) {
    int i = blockIdx.x * blockDim.x + threadIdx.x;   // grid covers N*D exactly
    float f = q[i];
    unsigned short h = f2bf_rn(f);
    float r = f - bf2f(h);
    qhi[i] = h;
    qlo[i] = f2bf_rn(r);
}

// ---------------------------------------------------------------------------
// Kernel 2: per-row squared norms (one wave per row).
// ---------------------------------------------------------------------------
__global__ void rl_sq_kernel(const float* __restrict__ q, float* __restrict__ sq) {
    int lane = threadIdx.x & 31;
    int wave = threadIdx.x >> 5;
    int row  = blockIdx.x * (blockDim.x >> 5) + wave;
    const float* r = q + (size_t)row * DIM;
    float acc = 0.f;
    #pragma unroll
    for (int k = lane; k < DIM; k += 32) { float v = r[k]; acc += v * v; }
    #pragma unroll
    for (int m = 1; m < 32; m <<= 1) acc += __shfl_xor(acc, m, 32);
    if (lane == 0) sq[row] = acc;
}

// ---------------------------------------------------------------------------
// Kernel 3: fused Gram-WMMA + masked argmax mining.
// A tile resident in VGPRs per wave; B tiles staged to LDS once per WG with
// double-buffered async global->LDS DMA (ASYNCcnt), consumed via ds_load_b128.
// ---------------------------------------------------------------------------
__global__ void __launch_bounds__(MINE_THREADS, 1)
rl_mine_kernel(const __bf16* __restrict__ qhi, const __bf16* __restrict__ qlo,
               const float* __restrict__ sq, const int* __restrict__ targets,
               float* __restrict__ posval, int* __restrict__ posidx,
               float* __restrict__ negval, int* __restrict__ negidx) {
    // LDS: [buffer][plane][row][k]
    __shared__ __align__(16) __bf16 sbuf[2][2][16][DIM];

    const int lane  = threadIdx.x & 31;
    const int wave  = threadIdx.x >> 5;
    const int chunk = blockIdx.y;
    const int rowbase = blockIdx.x * (MINE_WAVES * 16) + wave * 16;
    const int m     = lane & 15;        // A: row-in-tile; B/C: column-in-tile
    const int khalf = lane >> 4;        // K sub-offset selector per lane half

    const unsigned sbase = (unsigned)(unsigned long long)&sbuf[0][0][0][0];

    // ---- Load resident A tile (rows rowbase..rowbase+15), hi & lo planes.
    v16bf a_hi[NUM_KBLK], a_lo[NUM_KBLK];
    {
        const __bf16* arh = qhi + (size_t)(rowbase + m) * DIM;
        const __bf16* arl = qlo + (size_t)(rowbase + m) * DIM;
        #pragma unroll
        for (int kb = 0; kb < NUM_KBLK; ++kb) {
            int k0 = kb * 32 + khalf * 8;
            v8bf h0 = *(const v8bf*)(arh + k0);
            v8bf h1 = *(const v8bf*)(arh + k0 + 16);
            a_hi[kb] = __builtin_shufflevector(h0, h1, 0,1,2,3,4,5,6,7,8,9,10,11,12,13,14,15);
            v8bf l0 = *(const v8bf*)(arl + k0);
            v8bf l1 = *(const v8bf*)(arl + k0 + 16);
            a_lo[kb] = __builtin_shufflevector(l0, l1, 0,1,2,3,4,5,6,7,8,9,10,11,12,13,14,15);
        }
    }

    // ---- Per-C-slot row metadata (C layout: slot v -> row v + 8*khalf).
    float sqi[8]; int ti[8];
    #pragma unroll
    for (int v = 0; v < 8; ++v) {
        int r = rowbase + v + 8 * khalf;
        sqi[v] = sq[r];
        ti[v]  = targets[r];
    }

    const float NEG_INF = -__builtin_inff();
    float bpv[8], bnv[8]; int bpi[8], bni[8];
    #pragma unroll
    for (int v = 0; v < 8; ++v) {
        bpv[v] = NEG_INF; bnv[v] = NEG_INF;
        bpi[v] = 0x7fffffff; bni[v] = 0x7fffffff;
    }

    const int jbeg   = chunk * CHUNK;
    const int ntiles = CHUNK / 16;

    // ---- Async stage of one B tile (16 rows, hi+lo planes) into LDS buffer.
    // 1024 x 16B chunks, 8 per thread; one async DMA instruction per chunk.
    auto issue_tile = [&](int j0, int buf) {
        #pragma unroll
        for (int i = 0; i < CHUNKS_PER_THREAD; ++i) {
            int idx = (int)threadIdx.x + i * MINE_THREADS;    // 0..1023
            int p   = idx >> 9;                               // plane
            int rem = idx & 511;
            int r   = rem >> 5;                               // row in tile
            int cc  = rem & 31;                               // 8-elem chunk in row
            const __bf16* src = (p ? qlo : qhi) + (size_t)(j0 + r) * DIM + cc * 8;
            unsigned dst = sbase + (unsigned)(buf * TILE_BYTES)
                         + (unsigned)(((p * 16 + r) * DIM + cc * 8) * 2);
            asm volatile("global_load_async_to_lds_b128 %0, %1, off"
                         :: "v"(dst), "v"((unsigned long long)src) : "memory");
        }
    };

    issue_tile(jbeg, 0);   // prologue: prefetch first tile

    for (int t = 0; t < ntiles; ++t) {
        const int j0 = jbeg + t * 16;
        if (t + 1 < ntiles) {
            issue_tile(j0 + 16, (t + 1) & 1);
            asm volatile("s_wait_asynccnt 0x8" ::: "memory");  // tile t resident
        } else {
            asm volatile("s_wait_asynccnt 0x0" ::: "memory");
        }
        __syncthreads();   // all waves' DMA for tile t visible

        const int buf  = t & 1;
        const int jcol = j0 + m;                  // this lane's column in C

        v8f c = {0.f, 0.f, 0.f, 0.f, 0.f, 0.f, 0.f, 0.f};
        #pragma unroll
        for (int kb = 0; kb < NUM_KBLK; ++kb) {
            int k0 = kb * 32 + khalf * 8;
            v8bf h0 = *(const v8bf*)&sbuf[buf][0][m][k0];
            v8bf h1 = *(const v8bf*)&sbuf[buf][0][m][k0 + 16];
            v16bf b_hi = __builtin_shufflevector(h0, h1, 0,1,2,3,4,5,6,7,8,9,10,11,12,13,14,15);
            v8bf l0 = *(const v8bf*)&sbuf[buf][1][m][k0];
            v8bf l1 = *(const v8bf*)&sbuf[buf][1][m][k0 + 16];
            v16bf b_lo = __builtin_shufflevector(l0, l1, 0,1,2,3,4,5,6,7,8,9,10,11,12,13,14,15);
            // bf16x3 split: hi*hi + hi*lo + lo*hi (lo*lo negligible)
            c = __builtin_amdgcn_wmma_f32_16x16x32_bf16(false, a_hi[kb], false, b_hi,
                                                        (short)0, c, false, false);
            c = __builtin_amdgcn_wmma_f32_16x16x32_bf16(false, a_hi[kb], false, b_lo,
                                                        (short)0, c, false, false);
            c = __builtin_amdgcn_wmma_f32_16x16x32_bf16(false, a_lo[kb], false, b_hi,
                                                        (short)0, c, false, false);
        }

        const float sqj = sq[jcol];
        const int   tj  = targets[jcol];
        #pragma unroll
        for (int v = 0; v < 8; ++v) {
            int   irow = rowbase + v + 8 * khalf;
            float d    = sqi[v] + sqj - 2.0f * c[v];
            bool  diag = (irow == jcol);
            bool  same = (ti[v] == tj);
            // reference: pos = dist*same (0 where diff class), diag = -inf
            float pv = diag ? NEG_INF : (same ? d : 0.0f);
            float nv = diag ? NEG_INF : (same ? 0.0f : d);
            if (pv > bpv[v]) { bpv[v] = pv; bpi[v] = jcol; }   // strict > keeps
            if (nv > bnv[v]) { bnv[v] = nv; bni[v] = jcol; }   // first max index
        }

        __syncthreads();   // all waves done reading buf before it is re-filled
    }

    // ---- Cross-lane argmax reduction within each 16-lane half.
    #pragma unroll
    for (int v = 0; v < 8; ++v) {
        float pv = bpv[v]; int pi = bpi[v];
        float nv = bnv[v]; int ni = bni[v];
        #pragma unroll
        for (int msk = 1; msk < 16; msk <<= 1) {
            float opv = __shfl_xor(pv, msk, 32); int opi = __shfl_xor(pi, msk, 32);
            if (opv > pv || (opv == pv && opi < pi)) { pv = opv; pi = opi; }
            float onv = __shfl_xor(nv, msk, 32); int oni = __shfl_xor(ni, msk, 32);
            if (onv > nv || (onv == nv && oni < ni)) { nv = onv; ni = oni; }
        }
        if ((lane & 15) == 0) {
            int r = rowbase + v + 8 * khalf;
            posval[r * CSPLIT + chunk] = pv;
            posidx[r * CSPLIT + chunk] = pi;
            negval[r * CSPLIT + chunk] = nv;
            negidx[r * CSPLIT + chunk] = ni;
        }
    }
}

// ---------------------------------------------------------------------------
// Kernel 4: merge chunk candidates, exact fp32 re-score, per-row hinge loss.
// ---------------------------------------------------------------------------
__global__ void rl_finalize_kernel(const float* __restrict__ q,
                                   const float* __restrict__ posval, const int* __restrict__ posidx,
                                   const float* __restrict__ negval, const int* __restrict__ negidx,
                                   float* __restrict__ rowloss) {
    int lane = threadIdx.x & 31;
    int wave = threadIdx.x >> 5;
    int row  = blockIdx.x * (blockDim.x >> 5) + wave;

    const float NEG_INF = -__builtin_inff();
    float bpv = NEG_INF, bnv = NEG_INF;
    int bpi = 0, bni = 0;
    #pragma unroll
    for (int c = 0; c < CSPLIT; ++c) {           // ascending j-range order
        float pv = posval[row * CSPLIT + c]; int pi = posidx[row * CSPLIT + c];
        if (pv > bpv || (pv == bpv && pi < bpi)) { bpv = pv; bpi = pi; }
        float nv = negval[row * CSPLIT + c]; int ni = negidx[row * CSPLIT + c];
        if (nv > bnv || (nv == bnv && ni < bni)) { bnv = nv; bni = ni; }
    }

    const float* qi = q + (size_t)row * DIM;
    const float* qp = q + (size_t)bpi * DIM;
    const float* qn = q + (size_t)bni * DIM;
    float sp = 0.f, sn = 0.f;
    #pragma unroll
    for (int k = lane; k < DIM; k += 32) {
        float a = qi[k] - qp[k]; sp += a * a;
        float b = qi[k] - qn[k]; sn += b * b;
    }
    #pragma unroll
    for (int msk = 1; msk < 32; msk <<= 1) {
        sp += __shfl_xor(sp, msk, 32);
        sn += __shfl_xor(sn, msk, 32);
    }
    if (lane == 0) rowloss[row] = fmaxf(0.f, DELTA - sp + sn);
}

// ---------------------------------------------------------------------------
// Kernel 5: deterministic tree reduction -> mean loss.
// ---------------------------------------------------------------------------
__global__ void rl_reduce_kernel(const float* __restrict__ rowloss, float* __restrict__ out) {
    __shared__ float s[256];
    float acc = 0.f;
    for (int i = threadIdx.x; i < NROWS; i += 256) acc += rowloss[i];
    s[threadIdx.x] = acc;
    __syncthreads();
    for (int st = 128; st > 0; st >>= 1) {
        if ((int)threadIdx.x < st) s[threadIdx.x] += s[threadIdx.x + st];
        __syncthreads();
    }
    if (threadIdx.x == 0) out[0] = s[0] / (float)NROWS;
}

// ---------------------------------------------------------------------------
// Launch
// ---------------------------------------------------------------------------
extern "C" void kernel_launch(void* const* d_in, const int* in_sizes, int n_in,
                              void* d_out, int out_size, void* d_ws, size_t ws_size,
                              hipStream_t stream) {
    (void)in_sizes; (void)n_in; (void)out_size; (void)ws_size;
    const float* q       = (const float*)d_in[0];
    const int*   targets = (const int*)d_in[1];
    float*       out     = (float*)d_out;

    // Workspace layout (all offsets 256B-aligned).
    char* ws = (char*)d_ws;
    size_t off = 0;
    unsigned short* qhi = (unsigned short*)(ws + off); off += (size_t)NROWS * DIM * 2;  // 4 MB
    unsigned short* qlo = (unsigned short*)(ws + off); off += (size_t)NROWS * DIM * 2;  // 4 MB
    float* sq      = (float*)(ws + off); off += (size_t)NROWS * 4;
    float* posval  = (float*)(ws + off); off += (size_t)NROWS * CSPLIT * 4;
    int*   posidx  = (int*)  (ws + off); off += (size_t)NROWS * CSPLIT * 4;
    float* negval  = (float*)(ws + off); off += (size_t)NROWS * CSPLIT * 4;
    int*   negidx  = (int*)  (ws + off); off += (size_t)NROWS * CSPLIT * 4;
    float* rowloss = (float*)(ws + off); off += (size_t)NROWS * 4;

    // 1) bf16 hi/lo split.
    rl_split_kernel<<<(NROWS * DIM) / 256, 256, 0, stream>>>(q, qhi, qlo);
    // 2) row squared norms.
    rl_sq_kernel<<<NROWS / 8, 256, 0, stream>>>(q, sq);
    // 3) fused WMMA Gram + mining (async-LDS double-buffered B).
    dim3 mgrid(NROWS / (MINE_WAVES * 16), CSPLIT);
    rl_mine_kernel<<<mgrid, MINE_THREADS, 0, stream>>>(
        (const __bf16*)qhi, (const __bf16*)qlo, sq, targets,
        posval, posidx, negval, negidx);
    // 4) merge + exact rescoring + hinge.
    rl_finalize_kernel<<<NROWS / 8, 256, 0, stream>>>(q, posval, posidx, negval, negidx, rowloss);
    // 5) deterministic mean.
    rl_reduce_kernel<<<1, 256, 0, stream>>>(rowloss, out);
}
